// SO8RotationGate_45483703665178
// MI455X (gfx1250) — matrix operand order repaired and checked by reading, not compiled, not run on hardware
//
#include <hip/hip_runtime.h>
#include <hip/hip_bf16.h>

// SO8RotationGate for MI455X (gfx1250), wave32 + WMMA bf16.
// B=4, N=8192, D=1024, H=8, head_dim=128.

typedef __attribute__((ext_vector_type(16))) __bf16 bf16x16;
typedef __attribute__((ext_vector_type(8)))  __bf16 bf16x8;
typedef __attribute__((ext_vector_type(8)))  float  f32x8;

#define D_DIM  1024
#define N_SEQ  8192
#define B_DIM  4
#define H_DIM  8
#define TK     64          // K tile staged in LDS (2 WMMA k-steps of 32)
#define LDSK   (TK + 8)    // 72 bf16 = 144 B stride: 16B-aligned, conflict-free

__device__ __forceinline__ __bf16 f2bf(float f) {
    // round-to-nearest-even fp32 -> bf16
    unsigned u = __builtin_bit_cast(unsigned, f);
    unsigned r = u + 0x7FFFu + ((u >> 16) & 1u);
    unsigned short h = (unsigned short)(r >> 16);
    return __builtin_bit_cast(__bf16, h);
}

__global__ __launch_bounds__(256)
void so8_gemm_rot_kernel(const float* __restrict__ x,
                         const float* __restrict__ W,
                         const float* __restrict__ bias,
                         const float* __restrict__ rot_scale,
                         const float* __restrict__ angles,
                         float* __restrict__ out,
                         float* __restrict__ norm_ws)
{
    __shared__ __bf16 As[128][LDSK];     // x tile   (rows = M, cols = K)
    __shared__ __bf16 Bs[128][LDSK];     // W tile   (rows = N, cols = K)
    __shared__ float  Chead[128][8];     // head8 columns (x_proj + bias)

    const int tid    = threadIdx.x;
    const int lane   = tid & 31;
    const int wv     = tid >> 5;                 // 8 waves
    const int mrow   = blockIdx.x * 128;         // global row base (B*N flat)
    const int head   = blockIdx.y;               // 0..7
    const int hbase  = head * 128;               // global col base

    const int wm     = (wv >> 1) * 32;           // wave M offset in tile
    const int wn     = (wv & 1) * 64;            // wave N offset in tile
    const int laneN  = lane & 15;
    const int laneHi = (lane >> 4) & 1;

    f32x8 acc[2][4];
    const f32x8 zero = {0.f,0.f,0.f,0.f,0.f,0.f,0.f,0.f};
    #pragma unroll
    for (int i = 0; i < 2; ++i)
        #pragma unroll
        for (int j = 0; j < 4; ++j) acc[i][j] = zero;

    // cooperative tile-load coords: 2 threads per row, 32 floats each
    const int lr = tid >> 1;
    const int lc = (tid & 1) * 32;
    const float* xrow = x + (size_t)(mrow + lr) * D_DIM + lc;
    const float* wrow = W + (size_t)(hbase + lr) * D_DIM + lc;

    for (int kb = 0; kb < D_DIM; kb += TK) {
        if (kb + TK < D_DIM) {                       // prefetch next K tile
            __builtin_prefetch(xrow + kb + TK, 0, 3);
            __builtin_prefetch(wrow + kb + TK, 0, 3);
        }
        // ---- global fp32 -> bf16 -> LDS (b128 stores) ----
        #pragma unroll
        for (int cc = 0; cc < 4; ++cc) {
            float4 a0 = *(const float4*)(xrow + kb + cc * 8);
            float4 a1 = *(const float4*)(xrow + kb + cc * 8 + 4);
            bf16x8 av;
            av[0]=f2bf(a0.x); av[1]=f2bf(a0.y); av[2]=f2bf(a0.z); av[3]=f2bf(a0.w);
            av[4]=f2bf(a1.x); av[5]=f2bf(a1.y); av[6]=f2bf(a1.z); av[7]=f2bf(a1.w);
            *(bf16x8*)&As[lr][lc + cc * 8] = av;

            float4 b0 = *(const float4*)(wrow + kb + cc * 8);
            float4 b1 = *(const float4*)(wrow + kb + cc * 8 + 4);
            bf16x8 bv;
            bv[0]=f2bf(b0.x); bv[1]=f2bf(b0.y); bv[2]=f2bf(b0.z); bv[3]=f2bf(b0.w);
            bv[4]=f2bf(b1.x); bv[5]=f2bf(b1.y); bv[6]=f2bf(b1.z); bv[7]=f2bf(b1.w);
            *(bf16x8*)&Bs[lr][lc + cc * 8] = bv;
        }
        __syncthreads();

        // ---- 2 WMMA k-steps of 32 ----
        #pragma unroll
        for (int kk = 0; kk < TK; kk += 32) {
            bf16x16 afr[2];
            #pragma unroll
            for (int fm = 0; fm < 2; ++fm) {
                // A 16x32 bf16 layout: lanes<16 hold K 0-7 & 16-23, lanes>=16 hold 8-15 & 24-31
                const int row = wm + fm * 16 + laneN;
                const int ka  = kk + (laneHi ? 8 : 0);
                bf16x8 lo = *(const bf16x8*)&As[row][ka];
                bf16x8 hi = *(const bf16x8*)&As[row][ka + 16];
                afr[fm] = __builtin_shufflevector(lo, hi, 0,1,2,3,4,5,6,7,8,9,10,11,12,13,14,15);
            }
            bf16x16 bfr[4];
            #pragma unroll
            for (int fn = 0; fn < 4; ++fn) {
                // B 32x16 bf16 layout: lane = N; lanes<16 hold K 0-7 & 16-23, lanes>=16 hold 8-15 & 24-31
                const int col = wn + fn * 16 + laneN;
                const int kv  = kk + (laneHi ? 8 : 0);
                bf16x8 lo = *(const bf16x8*)&Bs[col][kv];
                bf16x8 hi = *(const bf16x8*)&Bs[col][kv + 16];
                bfr[fn] = __builtin_shufflevector(lo, hi, 0,1,2,3,4,5,6,7,8,9,10,11,12,13,14,15);
            }
            #pragma unroll
            for (int fm = 0; fm < 2; ++fm)
                #pragma unroll
                for (int fn = 0; fn < 4; ++fn)
                    acc[fm][fn] = __builtin_amdgcn_wmma_f32_16x16x32_bf16(
                        false, afr[fm], false, bfr[fn],
                        (short)0, acc[fm][fn], false, false);
        }
        __syncthreads();
    }

    // ---- epilogue: bias, residual, stash head8 columns to LDS ----
    #pragma unroll
    for (int fn = 0; fn < 4; ++fn) {
        const int colt = wn + fn * 16 + laneN;            // col within tile
        const float bv = bias[hbase + colt];
        #pragma unroll
        for (int fm = 0; fm < 2; ++fm) {
            const int rowb = wm + fm * 16 + (laneHi ? 8 : 0);
            #pragma unroll
            for (int j = 0; j < 8; ++j) {
                const float v    = acc[fm][fn][j] + bv;   // x_proj
                const int   rowt = rowb + j;
                if (colt < 8) {
                    Chead[rowt][colt] = v;                // rotated later
                } else {
                    const size_t gi = (size_t)(mrow + rowt) * D_DIM + hbase + colt;
                    out[gi] = v + x[gi];                  // rotated_x + x
                }
            }
        }
    }
    __syncthreads();

    // ---- SO8 rotation: one thread per row owns the full 8-vector ----
    if (tid < 128) {
        const int grow = mrow + tid;
        const int bi   = grow >> 13;                      // / 8192
        const int nn   = grow & (N_SEQ - 1);
        const float sc = rot_scale[0];
        const float* ap = angles + ((size_t)((bi * H_DIM + head) * N_SEQ + nn)) * 8;

        float r[8], o[8];
        #pragma unroll
        for (int j = 0; j < 8; ++j) { o[j] = Chead[tid][j]; r[j] = o[j]; }

        #pragma unroll
        for (int i = 0; i < 8; ++i) {
            const int p = 2 * (i & 3);
            const float th = ap[i] * sc;
            const float c = cosf(th), s = sinf(th), e = expf(th);
            const float xp = r[p], xp1 = r[p + 1];
            #pragma unroll
            for (int j = 0; j < 8; ++j) r[j] *= e;
            r[p]     = c * xp - s * xp1;
            r[p + 1] = s * xp + c * xp1;
        }
        float nrm = 0.f;
        #pragma unroll
        for (int j = 0; j < 8; ++j) { const float d = r[j] - o[j]; nrm += d * d; }
        nrm = sqrtf(nrm);

        const size_t gb = (size_t)grow * D_DIM + hbase;
        #pragma unroll
        for (int j = 0; j < 8; ++j) out[gb + j] = r[j] + x[gb + j];
        norm_ws[(size_t)(bi * H_DIM + head) * N_SEQ + nn] = nrm;  // per-head norm
    }
}

// deterministic reduction over heads: err[b,n] = mean_h norm[b,h,n]
__global__ __launch_bounds__(256)
void so8_err_reduce_kernel(const float* __restrict__ norm_ws,
                           float* __restrict__ err)
{
    const int i = blockIdx.x * blockDim.x + threadIdx.x;   // over B*N
    if (i >= B_DIM * N_SEQ) return;
    const int bi = i >> 13;
    const int nn = i & (N_SEQ - 1);
    float s = 0.f;
    #pragma unroll
    for (int h = 0; h < H_DIM; ++h)
        s += norm_ws[(size_t)(bi * H_DIM + h) * N_SEQ + nn];
    err[i] = s * 0.125f;
}

extern "C" void kernel_launch(void* const* d_in, const int* in_sizes, int n_in,
                              void* d_out, int out_size, void* d_ws, size_t ws_size,
                              hipStream_t stream) {
    const float* x      = (const float*)d_in[0];   // (B,N,D)
    const float* W      = (const float*)d_in[1];   // (D,D)
    const float* b      = (const float*)d_in[2];   // (D,)
    const float* rscale = (const float*)d_in[3];   // (1,)
    const float* ang    = (const float*)d_in[4];   // (B,H,N,8)

    float* out = (float*)d_out;                                    // (B,N,D) flat
    float* err = out + (size_t)B_DIM * N_SEQ * D_DIM;              // (B,N) flat
    float* nws = (float*)d_ws;  // needs B*H*N floats = 1 MB of scratch

    dim3 grid((B_DIM * N_SEQ) / 128, H_DIM);   // 256 x 8
    dim3 block(256);                            // 8 wave32 waves
    so8_gemm_rot_kernel<<<grid, block, 0, stream>>>(x, W, b, rscale, ang, out, nws);

    const int bn = B_DIM * N_SEQ;
    so8_err_reduce_kernel<<<(bn + 255) / 256, 256, 0, stream>>>(nws, err);
}
